// AttentionFusion_66692252172948
// MI455X (gfx1250) — compile-verified, matrix operand order checked by reading, not run on hardware
//
#include <hip/hip_runtime.h>
#include <math.h>
#include <stdint.h>

typedef __attribute__((ext_vector_type(16))) __bf16 v16bf;
typedef __attribute__((ext_vector_type(8)))  float  v8f;
typedef uint32_t u32x4 __attribute__((ext_vector_type(4)));
typedef int      i32x8 __attribute__((ext_vector_type(8)));
typedef int      i32x4 __attribute__((ext_vector_type(4)));

#define B_  8
#define S1_ 2048
#define S2_ 2048
#define D_  512
#define F_  256

__device__ __forceinline__ unsigned short f2bf(float x){
  union { float f; uint32_t u; } v; v.f = x;
  uint32_t r = (v.u + 0x7FFFu + ((v.u >> 16) & 1u)) >> 16;
  return (unsigned short)r;
}

// Load a 16x32 bf16 A-fragment (or matching B-fragment from a [n][k]
// K-contiguous LDS tile) per the CDNA5 16-bit A layout.
__device__ __forceinline__ v16bf ld_frag16(const unsigned short* p, int g){
  union { uint32_t u[8]; v16bf v; } f;
#pragma unroll
  for (int vv = 0; vv < 8; ++vv){
    int ko = ((vv < 4) ? 0 : 16) + ((vv & 3) << 1) + (g << 3);
    f.u[vv] = *(const uint32_t*)(p + ko);
  }
  return f.v;
}

__device__ __forceinline__ float redmax16(float v){
#pragma unroll
  for (int m = 8; m >= 1; m >>= 1) v = fmaxf(v, __shfl_xor(v, m, 16));
  return v;
}
__device__ __forceinline__ float redsum16(float v){
#pragma unroll
  for (int m = 8; m >= 1; m >>= 1) v += __shfl_xor(v, m, 16);
  return v;
}

// ---------------------------------------------------------------------------
// TDM: DMA a 2D bf16 tile (tile_d1 rows x tile_d0 elems) from global into LDS.
// data_size = 1 (2 bytes). Descriptor layout per CDNA5 ISA ch.8 (D# groups).
// Wave-level instruction (EXEC ignored); call from one wave only.
// ---------------------------------------------------------------------------
__device__ __forceinline__ void tdm_load_tile_2d(const void* lds_dst,
                                                 const void* global_src,
                                                 uint32_t tensor_d0,   // elems per row
                                                 uint32_t tensor_d1,   // rows in tensor
                                                 uint32_t tile_d0,     // tile width (elems)
                                                 uint32_t tile_d1,     // tile height (rows)
                                                 uint32_t d0_stride)   // row stride (elems)
{
  // Flat LDS pointers: the LDS address is the low 32 bits (ISA 10.2), so truncation gives
  // the LDS byte offset used by D#.lds_addr.
  uint32_t ldsaddr = (uint32_t)(uintptr_t)lds_dst;
  uint64_t ga      = (uint64_t)(uintptr_t)global_src;

  u32x4 g0;
  g0.x = 1u;                                   // count=1, user descriptor
  g0.y = ldsaddr;                              // lds_addr
  g0.z = (uint32_t)ga;                         // global_addr[31:0]
  g0.w = (uint32_t)((ga >> 32) & 0x1FFFFFFu)   // global_addr[56:32]
       | (2u << 30);                           // type=2 ("image")

  i32x8 g1;
  g1[0] = (int)(1u << 16);                                   // data_size=1 (2B)
  g1[1] = (int)((tensor_d0 & 0xFFFFu) << 16);                // tensor_dim0[15:0]
  g1[2] = (int)(((tensor_d0 >> 16) & 0xFFFFu)
              | ((tensor_d1 & 0xFFFFu) << 16));              // dim0 hi | dim1 lo
  g1[3] = (int)(((tensor_d1 >> 16) & 0xFFFFu)
              | ((tile_d0 & 0xFFFFu) << 16));                // dim1 hi | tile_dim0
  g1[4] = (int)(tile_d1 & 0xFFFFu);                          // tile_dim1 (tile_dim2=0)
  g1[5] = (int)d0_stride;                                    // tensor_dim0_stride[31:0]
  g1[6] = 0;                                                 // stride hi | dim1_stride lo
  g1[7] = 0;

  i32x4 z4 = {0, 0, 0, 0};
  i32x8 z8 = {0, 0, 0, 0, 0, 0, 0, 0};
  // clang-23 / therock form: (g0, g1, g2, g3, g_extra, cpol)
  __builtin_amdgcn_tensor_load_to_lds(g0, g1, z4, z4, z8, 0);
}

// ---------------------------------------------------------------------------
// GEMM: C[M,N] = A[M,Kd] * W[Kd,N] + bias.  A: f32 or bf16; C: bf16 or f32.
// Block = 256 threads (8 waves). Tile 64x64, BK=32. Wave grid 4(m) x 2(n).
// ---------------------------------------------------------------------------
template<bool A_BF16, bool OUT_BF16>
__global__ __launch_bounds__(256)
void gemm_bias_k(const void* __restrict__ Ain, const float* __restrict__ W,
                 const float* __restrict__ bias, void* __restrict__ Cout,
                 int M, int Kd, int N)
{
  __shared__ unsigned short As[64 * 32];   // [m][k]
  __shared__ unsigned short Ws[64 * 32];   // [n][k] (transposed)

  const int tid  = threadIdx.x;
  const int lane = tid & 31;
  const int wv   = tid >> 5;
  const int ln   = lane & 15;
  const int hf   = (lane >> 4) & 1;
  const int wm   = wv & 3;
  const int wn   = wv >> 2;
  const int m0   = blockIdx.y * 64;
  const int n0   = blockIdx.x * 64;

  v8f c0 = {}; v8f c1 = {};

  const int arow = tid >> 2;        // 0..63
  const int akof = (tid & 3) * 8;   // 0,8,16,24
  const int wk   = tid >> 3;        // 0..31
  const int wnb  = (tid & 7) * 8;   // 0..56

  for (int k0 = 0; k0 < Kd; k0 += 32){
    // ---- stage A tile (64 x 32) ----
    if (A_BF16){
      const unsigned short* Ab = (const unsigned short*)Ain;
      uint4 va = *(const uint4*)&Ab[(size_t)(m0 + arow) * Kd + k0 + akof];
      *(uint4*)&As[arow * 32 + akof] = va;
    } else {
      const float* Af = (const float*)Ain;
      const float4* pa = (const float4*)&Af[(size_t)(m0 + arow) * Kd + k0 + akof];
      float4 a0 = pa[0], a1 = pa[1];
      uint4 va;
      va.x = (uint32_t)f2bf(a0.x) | ((uint32_t)f2bf(a0.y) << 16);
      va.y = (uint32_t)f2bf(a0.z) | ((uint32_t)f2bf(a0.w) << 16);
      va.z = (uint32_t)f2bf(a1.x) | ((uint32_t)f2bf(a1.y) << 16);
      va.w = (uint32_t)f2bf(a1.z) | ((uint32_t)f2bf(a1.w) << 16);
      *(uint4*)&As[arow * 32 + akof] = va;
    }
    // ---- stage W tile (32 x 64) transposed into Ws[n][k] ----
    {
      const float4* pw = (const float4*)&W[(size_t)(k0 + wk) * N + n0 + wnb];
      float4 w0 = pw[0], w1 = pw[1];
      float wf[8] = {w0.x, w0.y, w0.z, w0.w, w1.x, w1.y, w1.z, w1.w};
#pragma unroll
      for (int j = 0; j < 8; ++j)
        Ws[(wnb + j) * 32 + wk] = f2bf(wf[j]);
    }
    __syncthreads();

    v16bf af = ld_frag16(&As[(wm * 16 + ln) * 32], hf);
    v16bf b0 = ld_frag16(&Ws[(wn * 32 +  0 + ln) * 32], hf);
    v16bf b1 = ld_frag16(&Ws[(wn * 32 + 16 + ln) * 32], hf);
    c0 = __builtin_amdgcn_wmma_f32_16x16x32_bf16(false, af, false, b0, (short)0, c0, false, false);
    c1 = __builtin_amdgcn_wmma_f32_16x16x32_bf16(false, af, false, b1, (short)0, c1, false, false);
    __syncthreads();
  }

#pragma unroll
  for (int nt = 0; nt < 2; ++nt){
#pragma unroll
    for (int r = 0; r < 8; ++r){
      int m = m0 + wm * 16 + r + 8 * hf;
      int n = n0 + wn * 32 + nt * 16 + ln;
      float val = (nt == 0 ? c0[r] : c1[r]) + bias[n];
      if (OUT_BF16) ((unsigned short*)Cout)[(size_t)m * N + n] = f2bf(val);
      else          ((float*)Cout)[(size_t)m * N + n] = val;
    }
  }
}

// ---------------------------------------------------------------------------
// Flash attention over bf16 Q/K/V. Block = 64 threads (2 waves), each wave
// owns 16 query rows; S2 processed in 32-col chunks. Q/K tiles staged by the
// Tensor Data Mover; V transposed manually during staging.
// ---------------------------------------------------------------------------
__global__ __launch_bounds__(64)
void attn_k(const unsigned short* __restrict__ Qb, const unsigned short* __restrict__ Kb,
            const unsigned short* __restrict__ Vb, unsigned short* __restrict__ AOb)
{
  __shared__ unsigned short Qs[32 * 256];   // [row][f]
  __shared__ unsigned short Ks[32 * 256];   // [t][f]
  __shared__ unsigned short Vt[256 * 32];   // [f][t]
  __shared__ unsigned short Ps[2][16 * 32]; // per-wave P bounce buffer

  const int tid  = threadIdx.x;
  const int lane = tid & 31;
  const int wv   = tid >> 5;        // 0..1
  const int ln   = lane & 15;
  const int hf   = (lane >> 4) & 1;

  const int b    = blockIdx.x >> 6;        // grid = B * (S1/32)
  const int s1_0 = (blockIdx.x & 63) * 32;

  // ---- stage Q tile (32 rows x 256) via TDM ----
  if (wv == 0){
    tdm_load_tile_2d(&Qs[0], &Qb[((size_t)b * S1_ + s1_0) * F_],
                     /*tensor_d0=*/F_, /*tensor_d1=*/B_ * S1_,
                     /*tile_d0=*/F_,  /*tile_d1=*/32, /*d0_stride=*/F_);
  }

  v8f zc = {};
  v8f acc[16];
#pragma unroll
  for (int nt = 0; nt < 16; ++nt) acc[nt] = zc;
  float mrow[8], lrow[8];
#pragma unroll
  for (int r = 0; r < 8; ++r){ mrow[r] = -3.0e38f; lrow[r] = 0.f; }

  const float scale = 0.0625f;  // 1/sqrt(F=256)

  for (int t0 = 0; t0 < S2_; t0 += 32){
    __syncthreads();   // previous chunk's LDS reads complete before restage

    // ---- stage K chunk (32 x 256) via TDM (wave 0 issues) ----
    if (wv == 0){
      tdm_load_tile_2d(&Ks[0], &Kb[((size_t)b * S2_ + t0) * F_],
                       F_, B_ * S2_, F_, 32, F_);
    }
    // ---- stage V chunk transposed (Vt[f][t]) while TDM runs ----
    {
      const int t  = tid >> 1;
      const int f0 = (tid & 1) * 128;
      const unsigned short* src = &Vb[((size_t)b * S2_ + t0 + t) * F_ + f0];
#pragma unroll
      for (int i = 0; i < 16; ++i){
        uint4 v = *(const uint4*)&src[i * 8];
        uint32_t w[4] = {v.x, v.y, v.z, v.w};
#pragma unroll
        for (int j = 0; j < 4; ++j){
          int f = f0 + i * 8 + j * 2;
          Vt[(f + 0) * 32 + t] = (unsigned short)(w[j] & 0xFFFFu);
          Vt[(f + 1) * 32 + t] = (unsigned short)(w[j] >> 16);
        }
      }
    }
    if (wv == 0) __builtin_amdgcn_s_wait_tensorcnt(0);  // Q (first iter) + K done
    __syncthreads();

    // ---- scores: S(16x32) = Q(16x256) * K^T(256x32) ----
    v8f c0 = zc, c1 = zc;
#pragma unroll
    for (int kk = 0; kk < 8; ++kk){
      v16bf qf = ld_frag16(&Qs[(wv * 16 + ln) * 256 + kk * 32], hf);
      v16bf k0 = ld_frag16(&Ks[( 0 + ln) * 256 + kk * 32], hf);
      v16bf k1 = ld_frag16(&Ks[(16 + ln) * 256 + kk * 32], hf);
      c0 = __builtin_amdgcn_wmma_f32_16x16x32_bf16(false, qf, false, k0, (short)0, c0, false, false);
      c1 = __builtin_amdgcn_wmma_f32_16x16x32_bf16(false, qf, false, k1, (short)0, c1, false, false);
    }

    // ---- online softmax ----
    float a_r[8];
#pragma unroll
    for (int r = 0; r < 8; ++r){
      float s0 = c0[r] * scale;
      float s1 = c1[r] * scale;
      float rm = redmax16(fmaxf(s0, s1));
      float nm = fmaxf(mrow[r], rm);
      float al = __expf(mrow[r] - nm);
      float p0 = __expf(s0 - nm);
      float p1 = __expf(s1 - nm);
      float rs = redsum16(p0 + p1);
      lrow[r] = lrow[r] * al + rs;
      mrow[r] = nm;
      a_r[r]  = al;
      int prow = r + 8 * hf;
      Ps[wv][prow * 32 +      ln] = f2bf(p0);
      Ps[wv][prow * 32 + 16 + ln] = f2bf(p1);
    }
#pragma unroll
    for (int nt = 0; nt < 16; ++nt){
#pragma unroll
      for (int r = 0; r < 8; ++r) acc[nt][r] *= a_r[r];
    }

    asm volatile("s_wait_dscnt 0" ::: "memory");  // per-wave P region ready

    // ---- O += P(16x32) * V(32x256) ----
    v16bf pf = ld_frag16(&Ps[wv][ln * 32], hf);
#pragma unroll
    for (int nt = 0; nt < 16; ++nt){
      v16bf vf = ld_frag16(&Vt[(nt * 16 + ln) * 32], hf);
      acc[nt] = __builtin_amdgcn_wmma_f32_16x16x32_bf16(false, pf, false, vf, (short)0, acc[nt], false, false);
    }
  }

  // ---- epilogue: normalize and store attn_out (bf16) ----
  float inv[8];
#pragma unroll
  for (int r = 0; r < 8; ++r) inv[r] = 1.0f / lrow[r];
#pragma unroll
  for (int nt = 0; nt < 16; ++nt){
#pragma unroll
    for (int r = 0; r < 8; ++r){
      int row = s1_0 + wv * 16 + r + 8 * hf;
      int f   = nt * 16 + ln;
      AOb[((size_t)b * S1_ + row) * F_ + f] = f2bf(acc[nt][r] * inv[r]);
    }
  }
}

// ---------------------------------------------------------------------------
extern "C" void kernel_launch(void* const* d_in, const int* in_sizes, int n_in,
                              void* d_out, int out_size, void* d_ws, size_t ws_size,
                              hipStream_t stream)
{
  (void)in_sizes; (void)n_in; (void)out_size; (void)ws_size;

  const float* feat1 = (const float*)d_in[0];
  const float* feat2 = (const float*)d_in[1];
  const float* Wq    = (const float*)d_in[2];
  const float* bq    = (const float*)d_in[3];
  const float* Wk    = (const float*)d_in[4];
  const float* bk    = (const float*)d_in[5];
  const float* Wv    = (const float*)d_in[6];
  const float* bv    = (const float*)d_in[7];
  const float* Wfc   = (const float*)d_in[8];
  const float* bfc   = (const float*)d_in[9];

  // workspace: Q,K,V,attn_out as bf16 -> 4 * B*S*F * 2B = 32 MB
  const size_t nQ = (size_t)B_ * S1_ * F_;
  unsigned short* Qb = (unsigned short*)d_ws;
  unsigned short* Kb = Qb + nQ;
  unsigned short* Vb = Kb + nQ;
  unsigned short* AO = Vb + nQ;

  dim3 gproj(F_ / 64, (B_ * S1_) / 64);   // (4, 256)
  gemm_bias_k<false, true><<<gproj, 256, 0, stream>>>(feat1, Wq, bq, Qb, B_ * S1_, D_, F_);
  gemm_bias_k<false, true><<<gproj, 256, 0, stream>>>(feat2, Wk, bk, Kb, B_ * S2_, D_, F_);
  gemm_bias_k<false, true><<<gproj, 256, 0, stream>>>(feat2, Wv, bv, Vb, B_ * S2_, D_, F_);

  attn_k<<<B_ * (S1_ / 32), 64, 0, stream>>>(Qb, Kb, Vb, AO);

  gemm_bias_k<true, false><<<dim3(F_ / 64, (B_ * S1_) / 64), 256, 0, stream>>>(
      AO, Wfc, bfc, d_out, B_ * S1_, F_, F_);
}
